// cross_set_score_69191923138676
// MI455X (gfx1250) — matrix operand order repaired and checked by reading, not compiled, not run on hardware
//
#include <hip/hip_runtime.h>
#include <hip/hip_bf16.h>

typedef __attribute__((ext_vector_type(16))) _Float16 v16h;
typedef __attribute__((ext_vector_type(8)))  _Float16 v8h;
typedef __attribute__((ext_vector_type(8)))  float    v8f;
typedef __attribute__((ext_vector_type(4)))  float    v4f;
typedef __attribute__((ext_vector_type(4)))  int      v4i;

typedef __attribute__((address_space(1))) v4i glob_v4i;   // global int4
typedef __attribute__((address_space(3))) v4i lds_v4i;    // LDS int4

#define NSET  16
#define MITEM 256
#define DIN   128
#define NHEAD 8
#define DHEAD 64
#define HCOLS (NHEAD * DHEAD) /* 512 */
#define ROWP  72              /* padded LDS row stride in f16 (144 B): bank-conflict free */
#define TPAD  24              /* transpose buffer row stride in f16 (48 B, 16B-aligned) */

#define SHUF16(a, b) __builtin_shufflevector((a), (b), 0,1,2,3,4,5,6,7,8,9,10,11,12,13,14,15)

#if defined(__has_builtin)
#  if __has_builtin(__builtin_amdgcn_global_load_async_to_lds_b128)
#    define HAVE_ASYNC_LDS 1
#  endif
#endif
#ifndef HAVE_ASYNC_LDS
#  define HAVE_ASYNC_LDS 0
#endif

// -------------------------------------------------------------------------
// Kernel 0: one-time repack of W1 (f32 [128][512]) into fp16 WMMA-B fragment
// order: W1h[frag=nt*4+kc][lane][e] so proj B-fragments are contiguous b128s.
// frag covers K = kc*32 + hgrp*16 + e, N = nt*16 + lmod.
// -------------------------------------------------------------------------
__global__ __launch_bounds__(256)
void w1_repack_kernel(const float* __restrict__ W1, _Float16* __restrict__ W1h) {
  const int gt   = blockIdx.x * 256 + threadIdx.x;   // 0..16383
  const int f    = gt >> 5;                          // fragment 0..511
  const int lane = gt & 31;
  const int nt   = f >> 2;
  const int kc   = f & 3;
  const int lmod = lane & 15;
  const int hgrp = lane >> 4;
  const float* src = W1 + (size_t)(kc * 32 + hgrp * 16) * HCOLS + nt * 16 + lmod;
  _Float16*    dst = W1h + ((size_t)f * 32 + lane) * 16;
#pragma unroll
  for (int e = 0; e < 16; ++e) dst[e] = (_Float16)src[(size_t)e * HCOLS];
}

// -------------------------------------------------------------------------
// Kernel 1: projection  h[pair][head][m][d] (fp16) = x[pair] @ W1
// One block per pair (i*16+j). 8 waves; wave handles mt in {2w, 2w+1},
// all 32 nt column tiles. 4 K-chunks of 32 per tile -> 4 WMMA per tile.
// Tile output is transposed through a private per-wave LDS buffer so each
// lane emits one coalesced global_store_b128 instead of 8 b16 stores.
// -------------------------------------------------------------------------
__global__ __launch_bounds__(256)
void proj_kernel(const float* __restrict__ x, const _Float16* __restrict__ W1h,
                 _Float16* __restrict__ hbuf) {
  __shared__ __align__(16) _Float16 xpose[8][16][TPAD];   // 6 KB, per-wave private

  const int pair = blockIdx.x;          // i*16 + j
  const int tid  = threadIdx.x;
  const int wave = tid >> 5;
  const int lane = tid & 31;
  const int lmod = lane & 15;
  const int hgrp = lane >> 4;           // 0 or 1 (half-wave group)
  const int trow = lane >> 1;           // 0..15: row this lane stores back
  const int thalf = lane & 1;           // which 8-col half of the row

  const float* xp = x + (size_t)pair * (MITEM * DIN);
  _Float16 (*tp)[TPAD] = xpose[wave];

  for (int mt2 = 0; mt2 < 2; ++mt2) {
    const int mt = wave * 2 + mt2;
    const int m  = mt * 16 + lmod;

    // Preload A fragments (x row m, 4 K-chunks of 32) via b128 loads + cvt.
    // A 16x32 layout (matches W1h B fragments): elems 0..15 -> K = kb + hgrp*16 + e
    v16h Afrag[4];
#pragma unroll
    for (int kc = 0; kc < 4; ++kc) {
      const float* ra = xp + (size_t)m * DIN + kc * 32 + hgrp * 16;
      v4f r0 = *(const v4f*)(ra);
      v4f r1 = *(const v4f*)(ra + 4);
      v4f r2 = *(const v4f*)(ra + 8);
      v4f r3 = *(const v4f*)(ra + 12);
      v16h a;
#pragma unroll
      for (int e = 0; e < 4; ++e) {
        a[e]      = (_Float16)r0[e];
        a[4 + e]  = (_Float16)r1[e];
        a[8 + e]  = (_Float16)r2[e];
        a[12 + e] = (_Float16)r3[e];
      }
      Afrag[kc] = a;
    }

    for (int nt = 0; nt < 32; ++nt) {
      v8f acc = {};
#pragma unroll
      for (int kc = 0; kc < 4; ++kc) {
        const _Float16* wb = W1h + ((size_t)(nt * 4 + kc) * 32 + lane) * 16;
        v8h b0 = *(const v8h*)(wb);
        v8h b1 = *(const v8h*)(wb + 8);
        v16h B = SHUF16(b0, b1);
        acc = __builtin_amdgcn_wmma_f32_16x16x32_f16(false, Afrag[kc], false, B,
                                                     (short)0, acc, false, false);
      }
      // C layout: lane elem r -> tile row (hgrp*8 + r), tile col lmod.
      // Transpose through per-wave LDS (in-order LDS pipe: no barrier needed),
      // then store one contiguous 16B row-half per lane.
#pragma unroll
      for (int r = 0; r < 8; ++r) tp[hgrp * 8 + r][lmod] = (_Float16)acc[r];

      v8h rowv = *(const v8h*)(&tp[trow][thalf * 8]);

      const int head = nt >> 2;
      const int dcol = ((nt & 3) << 4) + thalf * 8;
      _Float16* gp = hbuf + ((size_t)(pair * NHEAD + head) * MITEM
                             + (size_t)(mt * 16 + trow)) * DHEAD + dcol;
      *(v8h*)gp = rowv;
    }
  }
}

// -------------------------------------------------------------------------
// Kernel 2: scores. One block per (j,i,head) = blockIdx.x.
// S = h[j,i,head] (256x64) @ h[i,j,head]^T; LeakyReLU via sum/abs-sum trick:
// leaky(x/8) = 0.08125*x + 0.04375*|x|  (|x| is a free source modifier).
// B panel (32 KB) staged in LDS (async loads when available).
// -------------------------------------------------------------------------
__global__ __launch_bounds__(256)
void score_kernel(const _Float16* __restrict__ hbuf, float* __restrict__ sc) {
  __shared__ _Float16 ldsB[MITEM * ROWP];   // 36 KB
  __shared__ float red[8];

  const int b    = blockIdx.x;          // (j*16 + i)*8 + head
  const int head = b & 7;
  const int i    = (b >> 3) & 15;
  const int j    = b >> 7;
  const int tid  = threadIdx.x;
  const int wave = tid >> 5;
  const int lane = tid & 31;
  const int lmod = lane & 15;
  const int hgrp = lane >> 4;

  const _Float16* hA = hbuf + (size_t)((j * NSET + i) * NHEAD + head) * (MITEM * DHEAD);
  const _Float16* hB = hbuf + (size_t)((i * NSET + j) * NHEAD + head) * (MITEM * DHEAD);

  // ---- stage B panel: thread t copies row t (64 f16 = 8 x b128) ----
  {
    const _Float16* g = hB + (size_t)tid * DHEAD;
    _Float16* dst = ldsB + tid * ROWP;
#if HAVE_ASYNC_LDS
#pragma unroll
    for (int c = 0; c < 8; ++c)
      __builtin_amdgcn_global_load_async_to_lds_b128(
          (glob_v4i*)(g + c * 8),
          (lds_v4i*)(dst + c * 8), 0, 0);
    asm volatile("s_wait_asynccnt 0" ::: "memory");
#else
#pragma unroll
    for (int c = 0; c < 8; ++c)
      *(v8h*)(dst + c * 8) = *(const v8h*)(g + c * 8);
#endif
  }
  __syncthreads();

  float sumX = 0.f;   // plain sum of raw scores
  float sumA = 0.f;   // sum of |raw scores|

  for (int mt2 = 0; mt2 < 2; ++mt2) {
    const int mt = wave * 2 + mt2;
    const _Float16* pA = hA + (size_t)(mt * 16 + lmod) * DHEAD;
    // A fragments for the two K-chunks (d = 0..31 and 32..63), from global
    v8h a0 = *(const v8h*)(pA + hgrp * 16);
    v8h a1 = *(const v8h*)(pA + hgrp * 16 + 8);
    v16h A0 = SHUF16(a0, a1);
    v8h a2 = *(const v8h*)(pA + 32 + hgrp * 16);
    v8h a3 = *(const v8h*)(pA + 32 + hgrp * 16 + 8);
    v16h A1 = SHUF16(a2, a3);

    for (int nt = 0; nt < 16; ++nt) {
      // B[k][n] = h_ij[n][k]; row n = nt*16 + lmod, per-lane K run hgrp*16+e
      const _Float16* pB = ldsB + (size_t)(nt * 16 + lmod) * ROWP;
      v8h b0 = *(const v8h*)(pB + hgrp * 16);
      v8h b1 = *(const v8h*)(pB + hgrp * 16 + 8);
      v16h B0 = SHUF16(b0, b1);
      v8h b2 = *(const v8h*)(pB + 32 + hgrp * 16);
      v8h b3 = *(const v8h*)(pB + 32 + hgrp * 16 + 8);
      v16h B1 = SHUF16(b2, b3);

      v8f acc = {};
      acc = __builtin_amdgcn_wmma_f32_16x16x32_f16(false, A0, false, B0,
                                                   (short)0, acc, false, false);
      acc = __builtin_amdgcn_wmma_f32_16x16x32_f16(false, A1, false, B1,
                                                   (short)0, acc, false, false);
#pragma unroll
      for (int r = 0; r < 8; ++r) {
        sumX += acc[r];
        sumA += __builtin_fabsf(acc[r]);
      }
    }
  }

  // leaky(x/8) summed == 0.08125*sumX + 0.04375*sumA
  float waveSum = 0.08125f * sumX + 0.04375f * sumA;

  // wave32 reduction
#pragma unroll
  for (int off = 16; off > 0; off >>= 1)
    waveSum += __shfl_xor(waveSum, off, 32);

  if (lane == 0) red[wave] = waveSum;
  __syncthreads();
  if (tid == 0) {
    float s = 0.f;
#pragma unroll
    for (int w = 0; w < 8; ++w) s += red[w];
    sc[b] = s;
  }
}

// -------------------------------------------------------------------------
// Kernel 3: deterministic finalize. out[j*16+i] = (sum_h sc*W2[h]) / (nI*nJ)
// -------------------------------------------------------------------------
__global__ __launch_bounds__(256)
void finalize_kernel(const float* __restrict__ sc, const float* __restrict__ nItem,
                     const float* __restrict__ W2, float* __restrict__ out) {
  const int p = threadIdx.x;            // j*16 + i
  const int j = p >> 4;
  const int i = p & 15;
  float s = 0.f;
#pragma unroll
  for (int h = 0; h < NHEAD; ++h) s += sc[p * NHEAD + h] * W2[h];
  out[p] = s / (nItem[i] * nItem[j]);
}

extern "C" void kernel_launch(void* const* d_in, const int* in_sizes, int n_in,
                              void* d_out, int out_size, void* d_ws, size_t ws_size,
                              hipStream_t stream) {
  const float* x     = (const float*)d_in[0];
  const float* nItem = (const float*)d_in[1];
  const float* W1    = (const float*)d_in[2];
  const float* W2    = (const float*)d_in[3];

  _Float16* hbuf = (_Float16*)d_ws;                               // 64 MB fp16
  const size_t hBytes = (size_t)NSET * NSET * NHEAD * MITEM * DHEAD * sizeof(_Float16);
  float* sc = (float*)((char*)d_ws + hBytes);                     // 8 KB
  _Float16* W1h = (_Float16*)((char*)d_ws + hBytes + 8192);       // 512 KB

  w1_repack_kernel<<<64, 256, 0, stream>>>(W1, W1h);
  proj_kernel<<<NSET * NSET, 256, 0, stream>>>(x, W1h, hbuf);
  score_kernel<<<NSET * NSET * NHEAD, 256, 0, stream>>>(hbuf, sc);
  finalize_kernel<<<1, 256, 0, stream>>>(sc, nItem, W2, (float*)d_out);
}
